// DBTKT_84164179132922
// MI455X (gfx1250) — matrix operand order repaired and checked by reading, not compiled, not run on hardware
//
#include <hip/hip_runtime.h>
#include <hip/hip_bf16.h>

typedef __attribute__((ext_vector_type(16))) _Float16 v16h;
typedef __attribute__((ext_vector_type(8)))  float    v8f;

// Problem constants (from reference): B=32, LK=1024, N=2048, D=512, H=4, DK=128
#define PB   32
#define PLK  1024
#define PN   2048
#define PD   512
#define PH   4
#define PDK  128
#define P2D  1024
#define P3D  1536

// ---------------------------------------------------------------------------
// Kernel 1: convert a row-major f32 weight matrix (K x Nn) into f16 fragments
// pre-swizzled into the WMMA B-matrix lane layout (16x16x32 f16):
//   frag element e of lane L for tile (kt,nt) = W[kt*32 + (e>>3)*16 + (L>>4)*8 + (e&7)][nt*16 + (L&15)]
// stored at dst[(((kt*(Nn/16))+nt)*32 + L)*16 + e]  -> main kernel does 2x b128 loads.
// ---------------------------------------------------------------------------
__global__ void prep_weights(const float* __restrict__ W, _Float16* __restrict__ dst,
                             int K, int Nn) {
  int idx = blockIdx.x * blockDim.x + threadIdx.x;
  int total = K * Nn;
  if (idx >= total) return;
  int e    = idx & 15;
  int lane = (idx >> 4) & 31;
  int t    = idx >> 9;                 // tile index = kt*(Nn/16) + nt
  int nt   = t % (Nn >> 4);
  int kt   = t / (Nn >> 4);
  int k    = kt * 32 + (e >> 3) * 16 + (lane >> 4) * 8 + (e & 7);
  int n    = nt * 16 + (lane & 15);
  dst[idx] = (_Float16)W[(size_t)k * Nn + n];
}

// ---------------------------------------------------------------------------
// Kernel 2: per batch b, fold q-projection + head-split + Wk into:
//   u[b,h,e]  = sum_j Wk[e, h*128+j] * qh[b, h*128+j]      (e in [0,1024))
//   cb[b,h]   = qh[b,h,:] . bk_head                         (bias fold)
// so that scores[b,h,l] = (k[b,l,:] . u[b,h,:] + cb[b,h]) / sqrt(128).
// ---------------------------------------------------------------------------
__global__ void compute_u(const float* __restrict__ q,  const float* __restrict__ Wq,
                          const float* __restrict__ bq, const float* __restrict__ Wk,
                          const float* __restrict__ bk,
                          float* __restrict__ u, float* __restrict__ cb) {
  __shared__ float qrow[P2D];
  __shared__ float qh[PD];
  const int b = blockIdx.x;
  const int t = threadIdx.x;
  for (int i = t; i < P2D; i += 256) qrow[i] = q[(size_t)b * P2D + i];
  __syncthreads();
  for (int d = t; d < PD; d += 256) {
    float acc = bq[d];
    for (int e = 0; e < P2D; ++e) acc += qrow[e] * Wq[(size_t)e * PD + d];
    qh[d] = acc;
  }
  __syncthreads();
  if (t < PH) {
    float acc = 0.f;
    for (int j = 0; j < PDK; ++j) acc += qh[t * PDK + j] * bk[t * PDK + j];
    cb[b * PH + t] = acc;
  }
  for (int he = t; he < PH * P2D; he += 256) {
    int h = he >> 10;
    int e = he & (P2D - 1);
    const float* wr = Wk + (size_t)e * PD + h * PDK;
    const float* qv = qh + h * PDK;
    float acc = 0.f;
    for (int j = 0; j < PDK; ++j) acc += wr[j] * qv[j];
    u[((size_t)b * PH + h) * P2D + e] = acc;
  }
}

__device__ __forceinline__ float wave_reduce(float v) {
  for (int off = 16; off > 0; off >>= 1) v += __shfl_down(v, off, 32);
  return v;
}

// ---------------------------------------------------------------------------
// Kernel 3: scores[b,h,l] = (k[b,l,:].u[b,h,:] + cb[b,h]) * rsqrt(128).
// One wave per l-row; lanes stride the 1024-wide dot (coalesced), u in LDS.
// ---------------------------------------------------------------------------
__global__ void compute_scores(const float* __restrict__ k, const float* __restrict__ u,
                               const float* __restrict__ cb, float* __restrict__ scores) {
  __shared__ float us[PH * P2D];
  const int b = blockIdx.x >> 7;            // LK/8 = 128 blocks per batch
  const int t = threadIdx.x;
  for (int i = t; i < PH * P2D; i += 256) us[i] = u[(size_t)b * PH * P2D + i];
  __syncthreads();
  const int wave = t >> 5, lane = t & 31;
  const int l = ((blockIdx.x & 127) << 3) + wave;
  const float* kr = k + ((size_t)b * PLK + l) * P2D;
  float a0 = 0.f, a1 = 0.f, a2 = 0.f, a3 = 0.f;
  for (int j = lane; j < P2D; j += 32) {
    float kv = kr[j];
    a0 += kv * us[j];
    a1 += kv * us[P2D + j];
    a2 += kv * us[2 * P2D + j];
    a3 += kv * us[3 * P2D + j];
  }
  a0 = wave_reduce(a0); a1 = wave_reduce(a1);
  a2 = wave_reduce(a2); a3 = wave_reduce(a3);
  if (lane == 0) {
    const float sc = 0.08838834764831845f;  // 1/sqrt(128)
    float* sb = scores + (size_t)b * PH * PLK;
    const float* cbb = cb + b * PH;
    sb[0 * PLK + l] = (a0 + cbb[0]) * sc;
    sb[1 * PLK + l] = (a1 + cbb[1]) * sc;
    sb[2 * PLK + l] = (a2 + cbb[2]) * sc;
    sb[3 * PLK + l] = (a3 + cbb[3]) * sc;
  }
}

// ---------------------------------------------------------------------------
// Kernel 4: qe_score[b,h,n] = qe[b,n,:].scores[b,h,:]  (streams qe, 268 MB),
// emits for_correlation_weight = qe_score[:,0,:,0] and the sigmoid gate
//   g[b,h,n] = sigmoid(lam_h*cw[b,n] + (1-lam_h)*qe_score[b,h,n]).
// ---------------------------------------------------------------------------
__global__ void compute_gate(const float* __restrict__ qe, const float* __restrict__ scores,
                             const float* __restrict__ cw, const float* __restrict__ lambdas,
                             float* __restrict__ g, float* __restrict__ fcw) {
  __shared__ float ss[PH * PLK];
  const int b = blockIdx.x >> 8;            // N/8 = 256 blocks per batch
  const int t = threadIdx.x;
  for (int i = t; i < PH * PLK; i += 256) ss[i] = scores[(size_t)b * PH * PLK + i];
  __syncthreads();
  const int wave = t >> 5, lane = t & 31;
  const int n = ((blockIdx.x & 255) << 3) + wave;
  const float* qr = qe + ((size_t)b * PN + n) * PLK;
  float a0 = 0.f, a1 = 0.f, a2 = 0.f, a3 = 0.f;
  for (int j = lane; j < PLK; j += 32) {
    float qv = qr[j];
    a0 += qv * ss[j];
    a1 += qv * ss[PLK + j];
    a2 += qv * ss[2 * PLK + j];
    a3 += qv * ss[3 * PLK + j];
  }
  a0 = wave_reduce(a0); a1 = wave_reduce(a1);
  a2 = wave_reduce(a2); a3 = wave_reduce(a3);
  if (lane == 0) {
    fcw[(size_t)b * PN + n] = a0;
    const float c = cw[(size_t)b * PN + n];
    float qs[4] = {a0, a1, a2, a3};
    for (int h = 0; h < PH; ++h) {
      float lam = lambdas[h];
      float z = lam * c + (1.f - lam) * qs[h];
      g[((size_t)b * PH + h) * PN + n] = 1.f / (1.f + __expf(-z));
    }
  }
}

// ---------------------------------------------------------------------------
// Kernel 5: fused  vh = v_backward@Wv + bv  ->  gate  ->  out = (.)@Wo + bo
// Block: 32 rows of n (one batch), 256 threads = 8 waves.
//   wave w: rows m0 = 16*(w>>2), cols [128*(w&3), +128)  (= exactly head w&3)
//   -> 8 WMMA accum tiles of 16x16, K-loop 32 wide, all v_wmma_f32_16x16x32_f16.
// A staged f32->f16 through LDS; B pre-swizzled f16 fragments from L2.
// Gated intermediate kept in LDS (32x512 f16), never hits HBM.
// ---------------------------------------------------------------------------
__global__ void __launch_bounds__(256)
fused_out(const float* __restrict__ vb, const _Float16* __restrict__ WvH,
          const _Float16* __restrict__ WoH, const float* __restrict__ bv,
          const float* __restrict__ bo, const float* __restrict__ g,
          float* __restrict__ out) {
  extern __shared__ char smem[];
  _Float16* Ash = (_Float16*)smem;                              // [32][1536] 96 KB
  _Float16* Vsh = (_Float16*)(smem + 32 * P3D * 2);             // [32][512]  32 KB
  float*    gsh = (float*)(smem + 32 * P3D * 2 + 32 * PD * 2);  // [4][32]

  const int b  = blockIdx.x >> 6;     // N/32 = 64 blocks per batch
  const int n0 = (blockIdx.x & 63) << 5;
  const int t  = threadIdx.x;

  if (t < PH * 32) {
    int h = t >> 5, r = t & 31;
    gsh[h * 32 + r] = g[((size_t)b * PH + h) * PN + n0 + r];
  }
  // Coalesced load of the 32x1536 A tile, f32 -> f16 into LDS.
  const float* Asrc = vb + ((size_t)b * PN + n0) * P3D;
  for (int i = t; i < 32 * P3D; i += 256) Ash[i] = (_Float16)Asrc[i];
  __syncthreads();

  const int wave = t >> 5, lane = t & 31;
  const int mh = wave >> 2;           // 0/1 : which 16-row half
  const int nq = wave & 3;            // 128-col group == head index
  const int m0 = mh * 16;
  const int lm = lane & 15;           // A row-in-tile / B,C col-in-tile
  const int lg = lane >> 4;           // K-group selector

  // ---------------- GEMM1: vh_tile = A(32x1536) @ Wv(1536x512) ----------------
  v8f acc[8] = {};
  for (int kt = 0; kt < P3D / 32; ++kt) {
    union { v16h v; uint32_t u[8]; } A;
    const uint32_t* arp = (const uint32_t*)(Ash + (size_t)(m0 + lm) * P3D + kt * 32);
    const int kb = lg * 4;
    A.u[0] = arp[kb + 0]; A.u[1] = arp[kb + 1];
    A.u[2] = arp[kb + 2]; A.u[3] = arp[kb + 3];
    A.u[4] = arp[8 + kb + 0]; A.u[5] = arp[8 + kb + 1];
    A.u[6] = arp[8 + kb + 2]; A.u[7] = arp[8 + kb + 3];
#pragma unroll
    for (int ti = 0; ti < 8; ++ti) {
      const int ntile = nq * 8 + ti;
      v16h bf = *(const v16h*)(WvH + (((size_t)kt * 32 + ntile) * 32 + lane) * 16);
      acc[ti] = __builtin_amdgcn_wmma_f32_16x16x32_f16(
          false, A.v, false, bf, (short)0, acc[ti], false, false);
    }
  }

  // bias + sigmoid gate (head == nq), write f16 tile to LDS for GEMM2.
  float gv[8];
#pragma unroll
  for (int r = 0; r < 8; ++r) gv[r] = gsh[nq * 32 + m0 + r + lg * 8];
#pragma unroll
  for (int ti = 0; ti < 8; ++ti) {
    const int col = nq * 128 + ti * 16 + lm;
    const float bvv = bv[col];
#pragma unroll
    for (int r = 0; r < 8; ++r) {
      float v = (acc[ti][r] + bvv) * gv[r];
      Vsh[(size_t)(m0 + r + lg * 8) * PD + col] = (_Float16)v;
    }
  }
  __syncthreads();

  // ---------------- GEMM2: out_tile = gated(32x512) @ Wo(512x512) ----------------
  v8f acc2[8] = {};
  for (int kt = 0; kt < PD / 32; ++kt) {
    union { v16h v; uint32_t u[8]; } A;
    const uint32_t* arp = (const uint32_t*)(Vsh + (size_t)(m0 + lm) * PD + kt * 32);
    const int kb = lg * 4;
    A.u[0] = arp[kb + 0]; A.u[1] = arp[kb + 1];
    A.u[2] = arp[kb + 2]; A.u[3] = arp[kb + 3];
    A.u[4] = arp[8 + kb + 0]; A.u[5] = arp[8 + kb + 1];
    A.u[6] = arp[8 + kb + 2]; A.u[7] = arp[8 + kb + 3];
#pragma unroll
    for (int ti = 0; ti < 8; ++ti) {
      const int ntile = nq * 8 + ti;
      v16h bf = *(const v16h*)(WoH + (((size_t)kt * 32 + ntile) * 32 + lane) * 16);
      acc2[ti] = __builtin_amdgcn_wmma_f32_16x16x32_f16(
          false, A.v, false, bf, (short)0, acc2[ti], false, false);
    }
  }

  float* orow = out + ((size_t)b * PN + n0) * PD;
#pragma unroll
  for (int ti = 0; ti < 8; ++ti) {
    const int col = nq * 128 + ti * 16 + lm;
    const float bov = bo[col];
#pragma unroll
    for (int r = 0; r < 8; ++r) {
      orow[(size_t)(m0 + r + lg * 8) * PD + col] = acc2[ti][r] + bov;
    }
  }
}

// ---------------------------------------------------------------------------
extern "C" void kernel_launch(void* const* d_in, const int* in_sizes, int n_in,
                              void* d_out, int out_size, void* d_ws, size_t ws_size,
                              hipStream_t stream) {
  (void)in_sizes; (void)n_in; (void)out_size; (void)ws_size;
  const float* q       = (const float*)d_in[0];
  const float* k       = (const float*)d_in[1];
  const float* vb      = (const float*)d_in[2];
  const float* cw      = (const float*)d_in[3];
  const float* qe      = (const float*)d_in[4];
  const float* Wq      = (const float*)d_in[5];
  const float* bq      = (const float*)d_in[6];
  const float* Wk      = (const float*)d_in[7];
  const float* bk      = (const float*)d_in[8];
  const float* Wv      = (const float*)d_in[9];
  const float* bv      = (const float*)d_in[10];
  const float* Wo      = (const float*)d_in[11];
  const float* bo      = (const float*)d_in[12];
  const float* lambdas = (const float*)d_in[13];

  float* out = (float*)d_out;                      // output_backward (B,N,D)
  float* fcw = out + (size_t)PB * PN * PD;         // for_correlation_weight (B,N)

  char* w = (char*)d_ws;
  float* u      = (float*)w; w += (size_t)PB * PH * P2D * 4;
  float* cb     = (float*)w; w += (size_t)PB * PH * 4;
  float* scores = (float*)w; w += (size_t)PB * PH * PLK * 4;
  float* gbuf   = (float*)w; w += (size_t)PB * PH * PN * 4;
  _Float16* WvH = (_Float16*)w; w += (size_t)P3D * PD * 2;
  _Float16* WoH = (_Float16*)w; w += (size_t)PD * PD * 2;

  prep_weights<<<(P3D * PD + 255) / 256, 256, 0, stream>>>(Wv, WvH, P3D, PD);
  prep_weights<<<(PD * PD + 255) / 256, 256, 0, stream>>>(Wo, WoH, PD, PD);
  compute_u<<<PB, 256, 0, stream>>>(q, Wq, bq, Wk, bk, u, cb);
  compute_scores<<<PB * (PLK / 8), 256, 0, stream>>>(k, u, cb, scores);
  compute_gate<<<PB * (PN / 8), 256, 0, stream>>>(qe, scores, cw, lambdas, gbuf, fcw);

  size_t sm = (size_t)32 * P3D * 2 + (size_t)32 * PD * 2 + (size_t)PH * 32 * 4;
  fused_out<<<PB * (PN / 32), 256, sm, stream>>>(vb, WvH, WoH, bv, bo, gbuf, out);
}
